// RCNNPostProcess_83021717831992
// MI455X (gfx1250) — 8-bit matrix operand order repaired and checked by reading, compiled, not run on hardware
//
#include <hip/hip_runtime.h>
#include <hip/hip_bf16.h>

#define BB 4
#define RR 512
#define CC 80
#define KK 100
#define NMS_TH 0.3f
#define SCORE_TH 0.1f
#define NEGV (-1e9f)
#define DWH_CLIP 4.135166556742356f

typedef int v8i_t __attribute__((ext_vector_type(8)));

__device__ __forceinline__ bool iou_gt(float ax1, float ay1, float ax2, float ay2,
                                       float areaA,
                                       float bx1, float by1, float bx2, float by2,
                                       float areaB) {
    float lx = fmaxf(ax1, bx1), ly = fmaxf(ay1, by1);
    float rx = fminf(ax2, bx2), ry = fminf(ay2, by2);
    float w = fmaxf(rx - lx, 0.0f), h = fmaxf(ry - ly, 0.0f);
    float inter = w * h;
    return inter > NMS_TH * (areaA + areaB - inter + 1e-9f);
}

__device__ __forceinline__ float bcast_f(float v, int srclane) {
    return __int_as_float(__builtin_amdgcn_readlane(__float_as_int(v), srclane));
}
__device__ __forceinline__ int bcast_i(int v, int srclane) {
    return __builtin_amdgcn_readlane(v, srclane);
}

__global__ void __launch_bounds__(512)
rcnn_nms_kernel(const float* __restrict__ rois,
                const float* __restrict__ scores,
                const float* __restrict__ deltas,
                const int* __restrict__ im_hw,
                float* __restrict__ wsScores,
                float* __restrict__ wsBoxes) {
    __shared__ float sKey[RR];
    __shared__ int   sIdx[RR];
    __shared__ float boxTmp[RR * 4];
    __shared__ float bs4[RR * 4];
    __shared__ float ssArr[RR];
    __shared__ unsigned char keepArr[RR];

    const int b = blockIdx.x / CC;
    const int c = blockIdx.x % CC;
    const int r = threadIdx.x;

    {
        const float* rp = rois + (b * RR + r) * 4;
        float x1 = rp[0], y1 = rp[1], x2 = rp[2], y2 = rp[3];
        float w = x2 - x1, h = y2 - y1;
        float cx = x1 + 0.5f * w, cy = y1 + 0.5f * h;
        const float* dp = deltas + (size_t)(b * RR + r) * (CC + 1) * 4 + (c + 1) * 4;
        float dx = dp[0], dy = dp[1];
        float dw = fminf(dp[2], DWH_CLIP);
        float dh = fminf(dp[3], DWH_CLIP);
        float pcx = dx * w + cx, pcy = dy * h + cy;
        float pw = w * __expf(dw), ph = h * __expf(dh);
        float Hm1 = (float)im_hw[b * 2 + 0] - 1.0f;
        float Wm1 = (float)im_hw[b * 2 + 1] - 1.0f;
        float bx1 = fminf(fmaxf(pcx - 0.5f * pw, 0.0f), Wm1);
        float by1 = fminf(fmaxf(pcy - 0.5f * ph, 0.0f), Hm1);
        float bx2 = fminf(fmaxf(pcx + 0.5f * pw, 0.0f), Wm1);
        float by2 = fminf(fmaxf(pcy + 0.5f * ph, 0.0f), Hm1);
        boxTmp[r * 4 + 0] = bx1; boxTmp[r * 4 + 1] = by1;
        boxTmp[r * 4 + 2] = bx2; boxTmp[r * 4 + 3] = by2;
        sKey[r] = scores[(size_t)(b * RR + r) * (CC + 1) + (c + 1)];
        sIdx[r] = r;
        keepArr[r] = 0;
    }
    __syncthreads();

    for (int k = 2; k <= RR; k <<= 1) {
        for (int j = k >> 1; j > 0; j >>= 1) {
            int ixj = r ^ j;
            if (ixj > r) {
                float sA = sKey[r], sB = sKey[ixj];
                int iA = sIdx[r], iB = sIdx[ixj];
                bool before = (sA > sB) || (sA == sB && iA < iB);
                bool dirDesc = ((r & k) == 0);
                if (before != dirDesc) {
                    sKey[r] = sB; sKey[ixj] = sA;
                    sIdx[r] = iB; sIdx[ixj] = iA;
                }
            }
            __syncthreads();
        }
    }

    {
        int j = sIdx[r];
        ssArr[r] = sKey[r];
        bs4[r * 4 + 0] = boxTmp[j * 4 + 0];
        bs4[r * 4 + 1] = boxTmp[j * 4 + 1];
        bs4[r * 4 + 2] = boxTmp[j * 4 + 2];
        bs4[r * 4 + 3] = boxTmp[j * 4 + 3];
    }
    __syncthreads();

    if (threadIdx.x < 32) {
        const int lane = threadIdx.x;
        const int col  = lane & 15;
        const int half = lane >> 4;

        for (int tile = 0; tile < RR / 16; ++tile) {
            const int base = tile * 16;
            const int i = base + col;
            float cx1 = bs4[i * 4 + 0], cy1 = bs4[i * 4 + 1];
            float cx2 = bs4[i * 4 + 2], cy2 = bs4[i * 4 + 3];
            float cS  = ssArr[i];
            float areaC = (cx2 - cx1) * (cy2 - cy1);

            v8i_t acc = {0, 0, 0, 0, 0, 0, 0, 0};
            const int nblk = (base + 63) >> 6;
            for (int u = 0; u < nblk; ++u) {
                const int j0 = u * 64;
                v8i_t aF, bF;
#pragma unroll
                for (int v = 0; v < 8; ++v) {
                    int kb = ((v >> 1) << 4) + ((v & 1) << 2) + (half << 3);
                    unsigned wrd = 0;
#pragma unroll
                    for (int q = 0; q < 4; ++q) {
                        wrd |= ((unsigned)keepArr[j0 + kb + q]) << (8 * q);
                    }
                    aF[v] = (int)wrd;
                }
#pragma unroll
                for (int v = 0; v < 8; ++v) {
                    int kb = ((v >> 1) << 4) + ((v & 1) << 2) + (half << 3);
                    unsigned wrd = 0;
#pragma unroll
                    for (int q = 0; q < 4; ++q) {
                        int j = j0 + kb + q;
                        float jx1 = bs4[j * 4 + 0], jy1 = bs4[j * 4 + 1];
                        float jx2 = bs4[j * 4 + 2], jy2 = bs4[j * 4 + 3];
                        float areaJ = (jx2 - jx1) * (jy2 - jy1);
                        unsigned bit = iou_gt(jx1, jy1, jx2, jy2, areaJ,
                                              cx1, cy1, cx2, cy2, areaC) ? 1u : 0u;
                        wrd |= bit << (8 * q);
                    }
                    bF[v] = (int)wrd;
                }
                acc = __builtin_amdgcn_wmma_i32_16x16x64_iu8(
                    false, aF, false, bF, acc, false, false);
            }
            int supcnt = acc[0];

            int supTile = 0;
            unsigned myKeep = 0;
            for (int m = 0; m < 16; ++m) {
                int   sc_m = bcast_i(supcnt, m);
                float s_m  = bcast_f(cS, m);
                int   st_m = bcast_i(supTile, m);
                bool keep_m = (s_m > SCORE_TH) && (sc_m == 0) && (st_m == 0);
                if (keep_m) {
                    float mx1 = bcast_f(cx1, m), my1 = bcast_f(cy1, m);
                    float mx2 = bcast_f(cx2, m), my2 = bcast_f(cy2, m);
                    float areaM = (mx2 - mx1) * (my2 - my1);
                    bool sup = iou_gt(mx1, my1, mx2, my2, areaM,
                                      cx1, cy1, cx2, cy2, areaC);
                    if (sup && col > m) supTile = 1;
                }
                if (col == m) myKeep = keep_m ? 1u : 0u;
            }
            if (lane < 16) keepArr[base + lane] = (unsigned char)myKeep;
        }
    }
    __syncthreads();

    {
        size_t idx = ((size_t)b * CC + c) * RR + r;
        wsScores[idx] = keepArr[r] ? ssArr[r] : NEGV;
        wsBoxes[idx * 4 + 0] = bs4[r * 4 + 0];
        wsBoxes[idx * 4 + 1] = bs4[r * 4 + 1];
        wsBoxes[idx * 4 + 2] = bs4[r * 4 + 2];
        wsBoxes[idx * 4 + 3] = bs4[r * 4 + 3];
    }
}

__global__ void __launch_bounds__(1024)
rcnn_topk_kernel(float* __restrict__ wsScores,
                 const float* __restrict__ wsBoxes,
                 float* __restrict__ out) {
    __shared__ float rS[32];
    __shared__ int   rI[32];
    const int img = blockIdx.x;
    const int t = threadIdx.x;
    const int wave = t >> 5;
    const int lane = t & 31;
    const int N = CC * RR;
    float* sc = wsScores + (size_t)img * N;
    const float* bx = wsBoxes + (size_t)img * N * 4;

    for (int k = 0; k < KK; ++k) {
        float bestS = -3.0e38f;
        int bestI = 0;
        for (int i = t; i < N; i += 1024) {
            float v = sc[i];
            if (v > bestS) { bestS = v; bestI = i; }
        }
#pragma unroll
        for (int off = 16; off > 0; off >>= 1) {
            float so = __shfl_xor(bestS, off, 32);
            int   io = __shfl_xor(bestI, off, 32);
            if (so > bestS || (so == bestS && io < bestI)) { bestS = so; bestI = io; }
        }
        if (lane == 0) { rS[wave] = bestS; rI[wave] = bestI; }
        __syncthreads();
        if (wave == 0) {
            bestS = rS[lane];
            bestI = rI[lane];
#pragma unroll
            for (int off = 16; off > 0; off >>= 1) {
                float so = __shfl_xor(bestS, off, 32);
                int   io = __shfl_xor(bestI, off, 32);
                if (so > bestS || (so == bestS && io < bestI)) { bestS = so; bestI = io; }
            }
            if (lane == 0) {
                float* o = out + ((size_t)img * KK + k) * 6;
                if (bestS > NEGV * 0.5f) {
                    o[0] = bx[bestI * 4 + 0]; o[1] = bx[bestI * 4 + 1];
                    o[2] = bx[bestI * 4 + 2]; o[3] = bx[bestI * 4 + 3];
                    o[4] = bestS;             o[5] = (float)(bestI / RR);
                } else {
                    o[0] = 0.0f; o[1] = 0.0f; o[2] = 0.0f;
                    o[3] = 0.0f; o[4] = 0.0f; o[5] = -1.0f;
                }
                sc[bestI] = NEGV;
            }
        }
        __syncthreads();
    }
}

extern "C" void kernel_launch(void* const* d_in, const int* in_sizes, int n_in,
                              void* d_out, int out_size, void* d_ws, size_t ws_size,
                              hipStream_t stream) {
    const float* rois   = (const float*)d_in[0];
    const float* scores = (const float*)d_in[1];
    const float* deltas = (const float*)d_in[2];
    const int*   im_hw  = (const int*)d_in[3];
    float* out = (float*)d_out;

    float* wsScores = (float*)d_ws;
    float* wsBoxes  = wsScores + (size_t)BB * CC * RR;

    rcnn_nms_kernel<<<dim3(BB * CC), dim3(512), 0, stream>>>(
        rois, scores, deltas, im_hw, wsScores, wsBoxes);
    rcnn_topk_kernel<<<dim3(BB), dim3(1024), 0, stream>>>(
        wsScores, wsBoxes, out);
}
